// QTNet_8409545965899
// MI455X (gfx1250) — 8-bit matrix operand order repaired and checked by reading, compiled, not run on hardware
//
#include <hip/hip_runtime.h>
#include <hip/hip_bf16.h>
#include <stdint.h>

#define BATCH 16
#define CH    128
#define HH    64
#define WW    64
#define NPIX  (BATCH * HH * WW)
#define ACT_BYTES ((size_t)NPIX * CH)
#define WPK_BYTES ((size_t)CH * CH * 9)
#define NTILES    (CH / 16)
#define STAGE_BYTES (NTILES * 32 * 32)

typedef __attribute__((ext_vector_type(8))) int v8i;

__global__ void init_scalars_kernel(unsigned* wmax) {
    if (threadIdx.x < 2) wmax[threadIdx.x] = 0u;
}

__global__ void maxabs_kernel(const float* __restrict__ w, int n, unsigned* out) {
    int tid = blockIdx.x * blockDim.x + threadIdx.x;
    int stride = gridDim.x * blockDim.x;
    float m = 0.0f;
    for (int i = tid; i < n; i += stride) m = fmaxf(m, fabsf(w[i]));
    atomicMax(out, __float_as_uint(m));
}

__global__ void pack_weights_kernel(const float* __restrict__ w,
                                    const unsigned* __restrict__ maxbits,
                                    signed char* __restrict__ wpk) {
    int tid = blockIdx.x * blockDim.x + threadIdx.x;
    if (tid >= (int)WPK_BYTES) return;
    int j     = tid & 31;
    int lane  = (tid >> 5) & 31;
    int ntile = (tid >> 10) & 7;
    int chunk = (tid >> 13) & 1;
    int tap   = tid >> 14;
    int half  = lane >> 4;
    int n     = lane & 15;
    int k     = (j < 16) ? (half * 16 + j) : (32 + half * 16 + (j - 16));
    int cin   = chunk * 64 + k;
    int cout  = ntile * 16 + n;
    int ky = tap / 3, kx = tap % 3;
    float s = __uint_as_float(*maxbits) * (1.0f / 127.0f);
    float v = w[((size_t)(cout * CH + cin) * 3 + ky) * 3 + kx];
    v = fminf(fmaxf(v, -128.0f * s), 127.0f * s);
    wpk[tid] = (signed char)rintf(v / s);
}

__global__ void quant_input_kernel(const float* __restrict__ x,
                                   const float* __restrict__ s_p,
                                   signed char* __restrict__ out) {
    size_t i = (size_t)blockIdx.x * blockDim.x + threadIdx.x;
    if (i >= (size_t)BATCH * CH * HH * WW) return;
    int hw = (int)(i % (HH * WW));
    int c  = (int)((i / (HH * WW)) % CH);
    int b  = (int)(i / ((size_t)CH * HH * WW));
    float s = s_p[0];
    float v = x[i];
    v = fminf(fmaxf(v, -128.0f * s), 127.0f * s);
    out[((size_t)b * HH * WW + hw) * CH + c] = (signed char)rintf(v / s);
}

template <bool A_SIGNED, bool FINAL>
__global__ void conv_iu8_wmma_kernel(const signed char* __restrict__ actin,
                                     const signed char* __restrict__ wpk,
                                     const float* __restrict__ bias,
                                     const float* __restrict__ s_prev_p,
                                     const unsigned* __restrict__ wmax_bits,
                                     const float* __restrict__ s_act_p,
                                     void* __restrict__ outp) {
    __shared__ int4 smem[STAGE_BYTES / 16];

    const int lane  = threadIdx.x & 31;
    const int wid   = threadIdx.x >> 5;
    const int half  = lane >> 4;
    const int mrow  = lane & 15;

    const int p  = blockIdx.x * 128 + wid * 16 + mrow;
    const int b  = p / (HH * WW);
    const int rm = p % (HH * WW);
    const int oy = rm / WW;
    const int ox = rm % WW;

    v8i acc[NTILES];
#pragma unroll
    for (int nt = 0; nt < NTILES; ++nt) acc[nt] = (v8i){};

#pragma unroll 1
    for (int tap = 0; tap < 9; ++tap) {
        const int iy = oy + tap / 3 - 1;
        const int ix = ox + tap % 3 - 1;
        const bool valid = (iy >= 0) & (iy < HH) & (ix >= 0) & (ix < WW);
        const signed char* abase = actin + (size_t)((b * HH + iy) * WW + ix) * CH;

#pragma unroll 1
        for (int chunk = 0; chunk < 2; ++chunk) {
            const int round = tap * 2 + chunk;
            __syncthreads();
            {
                const int4* g = (const int4*)(wpk + (size_t)round * STAGE_BYTES
                                                  + (size_t)threadIdx.x * 32);
                smem[threadIdx.x * 2 + 0] = g[0];
                smem[threadIdx.x * 2 + 1] = g[1];
            }
            if (round < 17) {
                __builtin_prefetch(wpk + (size_t)(round + 1) * STAGE_BYTES
                                       + (size_t)threadIdx.x * 32, 0, 3);
            }
            __syncthreads();

            v8i a = {};
            if (valid) {
#pragma unroll
                for (int g = 0; g < 4; ++g) {
                    const uint2 q = *(const uint2*)(abase + chunk * 64 + (g >> 1) * 32 + half * 16 + (g & 1) * 8);
                    a[2 * g]     = (int)q.x;
                    a[2 * g + 1] = (int)q.y;
                }
            }

#pragma unroll
            for (int grp = 0; grp < 2; ++grp) {
                const int4* sbase = &smem[(grp * 4 * 32 + lane) * 2];
                v8i b0 = *(const v8i*)(sbase + 0 * 64);
                v8i b1 = *(const v8i*)(sbase + 1 * 64);
                v8i b2 = *(const v8i*)(sbase + 2 * 64);
                v8i b3 = *(const v8i*)(sbase + 3 * 64);
                asm volatile("" : "+v"(b0), "+v"(b1), "+v"(b2), "+v"(b3));
                acc[grp * 4 + 0] = __builtin_amdgcn_wmma_i32_16x16x64_iu8(
                    A_SIGNED, a, true, b0, acc[grp * 4 + 0], false, false);
                acc[grp * 4 + 1] = __builtin_amdgcn_wmma_i32_16x16x64_iu8(
                    A_SIGNED, a, true, b1, acc[grp * 4 + 1], false, false);
                acc[grp * 4 + 2] = __builtin_amdgcn_wmma_i32_16x16x64_iu8(
                    A_SIGNED, a, true, b2, acc[grp * 4 + 2], false, false);
                acc[grp * 4 + 3] = __builtin_amdgcn_wmma_i32_16x16x64_iu8(
                    A_SIGNED, a, true, b3, acc[grp * 4 + 3], false, false);
            }
        }
    }

    const float s_w    = __uint_as_float(*wmax_bits) * (1.0f / 127.0f);
    const float s_b    = s_prev_p[0] * s_w;
    const float inv_sb = 1.0f / s_b;
    const float s_a    = s_act_p[0];
    const float inv_sa = 1.0f / s_a;
    const float ymax   = 255.0f * s_a;

    float bqv[NTILES];
#pragma unroll
    for (int nt = 0; nt < NTILES; ++nt)
        bqv[nt] = rintf(bias[nt * 16 + (lane & 15)] * inv_sb);

#pragma unroll
    for (int r = 0; r < 8; ++r) {
        const int m  = r + 8 * half;
        const int po = blockIdx.x * 128 + wid * 16 + m;
        const int bb = po / (HH * WW);
        const int rr = po % (HH * WW);
        const int yy = rr / WW;
        const int xx = rr % WW;
        const size_t fbase = ((size_t)bb * CH * HH + (size_t)yy) * WW + xx;
        const size_t ubase = (size_t)po * CH;
#pragma unroll
        for (int nt = 0; nt < NTILES; ++nt) {
            const int n = nt * 16 + (lane & 15);
            float y = s_b * ((float)acc[nt][r] + bqv[nt]);
            y = fmaxf(y, 0.0f);
            float q = rintf(fminf(y, ymax) * inv_sa);
            q = fminf(fmaxf(q, 0.0f), 255.0f);
            if (FINAL) {
                ((float*)outp)[fbase + (size_t)n * (HH * WW)] = q * s_a;
            } else {
                ((unsigned char*)outp)[ubase + n] = (unsigned char)q;
            }
        }
    }
}

extern "C" void kernel_launch(void* const* d_in, const int* in_sizes, int n_in,
                              void* d_out, int out_size, void* d_ws, size_t ws_size,
                              hipStream_t stream) {
    const float* x    = (const float*)d_in[0];
    const float* w1   = (const float*)d_in[1];
    const float* b1   = (const float*)d_in[2];
    const float* w2   = (const float*)d_in[3];
    const float* b2   = (const float*)d_in[4];
    const float* s_in = (const float*)d_in[5];
    const float* s_a1 = (const float*)d_in[6];
    const float* s_a2 = (const float*)d_in[7];

    char* ws = (char*)d_ws;
    unsigned*      wmax = (unsigned*)ws;
    signed char*   act0 = (signed char*)(ws + 256);
    unsigned char* act1 = (unsigned char*)(ws + 256 + ACT_BYTES);
    signed char*   wp1  = (signed char*)(ws + 256 + 2 * ACT_BYTES);
    signed char*   wp2  = wp1 + WPK_BYTES;

    const int nW = (int)(CH * CH * 9);

    init_scalars_kernel<<<1, 32, 0, stream>>>(wmax);
    maxabs_kernel<<<128, 256, 0, stream>>>(w1, nW, wmax + 0);
    maxabs_kernel<<<128, 256, 0, stream>>>(w2, nW, wmax + 1);
    pack_weights_kernel<<<(int)((WPK_BYTES + 255) / 256), 256, 0, stream>>>(w1, wmax + 0, wp1);
    pack_weights_kernel<<<(int)((WPK_BYTES + 255) / 256), 256, 0, stream>>>(w2, wmax + 1, wp2);

    const size_t nIn = (size_t)BATCH * CH * HH * WW;
    quant_input_kernel<<<(int)((nIn + 255) / 256), 256, 0, stream>>>(x, s_in, act0);

    const int blocks = NPIX / 128;

    conv_iu8_wmma_kernel<true, false><<<blocks, 256, 0, stream>>>(
        act0, wp1, b1, s_in, wmax + 0, s_a1, (void*)act1);
    conv_iu8_wmma_kernel<false, true><<<blocks, 256, 0, stream>>>(
        (const signed char*)act1, wp2, b2, s_a1, wmax + 1, s_a2, d_out);
}
